// ACBlock_71356586656468
// MI455X (gfx1250) — compile-verified, matrix operand order checked by reading, not run on hardware
//
#include <hip/hip_runtime.h>

// ---------------------------------------------------------------------------
// Performer attention block for MI455X (gfx1250, wave32, WMMA 16x16x32 f16).
// All contractions (QKV, phi-projection, phi^T*V, phi*ctx, Wo, 3x3 conv as
// implicit GEMM) use v_wmma_f32_16x16x32_f16 (f16 operands, f32 accum).
// GEMM/conv tiles are staged with GLOBAL_LOAD_ASYNC_TO_LDS_B128 (ASYNCcnt)
// into double-buffered LDS. BatchNorm statistics stay in fp32.
// ---------------------------------------------------------------------------

typedef __attribute__((ext_vector_type(16))) _Float16 v16h;
typedef __attribute__((ext_vector_type(8)))  float    v8f;

#define BATCH   16
#define CHN     256
#define HDIM    56
#define HW      3136      // 56*56
#define NTOK    50176     // BATCH*HW
#define HEADS   8
#define DH      32
#define MF      110       // performer features
#define MFP     112       // padded to 7*16
#define FT      7         // feature tiles of 16
#define SPLIT_A 7         // token splits in ctx pass (3136/7 = 448 = 14*32)

union FragU { v16h h; unsigned int u[8]; };

__device__ __forceinline__ v8f zero8() {
  v8f z;
#pragma unroll
  for (int i = 0; i < 8; ++i) z[i] = 0.0f;
  return z;
}

// Per-lane f16 fragment load for WMMA 16x16x32 A/B operands.
// ISA 7.12.2: element pair v holds K0 = (v<4 ? 2v : 2v+8) + 8*half, K0+1.
__device__ __forceinline__ v16h load_frag(const _Float16* row, int halfsel) {
  FragU f;
#pragma unroll
  for (int v = 0; v < 8; ++v) {
    int k0 = (v < 4 ? 2 * v : 2 * v + 8) + 8 * halfsel;
    f.u[v] = *(const unsigned int*)(row + k0);
  }
  return f.h;
}

__device__ __forceinline__ v8f wmma_f16(v16h a, v16h b, v8f c) {
  return __builtin_amdgcn_wmma_f32_16x16x32_f16(false, a, false, b, (short)0, c,
                                                false, false);
}

// gfx1250 async copy global -> LDS, 16B per lane, tracked by ASYNCcnt.
__device__ __forceinline__ void async_b128(unsigned lds_off, const void* gaddr) {
  asm volatile("global_load_async_to_lds_b128 %0, %1, off"
               :: "v"(lds_off), "v"(gaddr) : "memory");
}
__device__ __forceinline__ void wait_async() {
  asm volatile("s_wait_asynccnt 0" ::: "memory");
}
__device__ __forceinline__ unsigned lds_addr(const void* p) {
  return (unsigned)(unsigned long long)p;
}

// ---------------------------------------------------------------------------
// Utility kernels
// ---------------------------------------------------------------------------

__global__ void zero_f32_kernel(float* __restrict__ p, int n) {
  int i = blockIdx.x * blockDim.x + threadIdx.x;
  if (i < n) p[i] = 0.0f;
}

// dT[n][k] = s[k][n], f16  (weights pre-transposed so GEMM B-tiles are
// contiguous 64B rows -> async B128 friendly)
__global__ void f32_to_f16_T_kernel(const float* __restrict__ s,
                                    _Float16* __restrict__ dT) {
  int i = blockIdx.x * blockDim.x + threadIdx.x;
  if (i >= CHN * CHN) return;
  int n = i >> 8, k = i & 255;
  dT[i] = (_Float16)s[(size_t)k * CHN + n];
}

// projH[f][d] = f<110 ? proj[f][d] * d^-0.25 : 0
__global__ void proj_convert_kernel(const float* __restrict__ proj,
                                    _Float16* __restrict__ projH) {
  int i = blockIdx.x * blockDim.x + threadIdx.x;
  if (i >= MFP * DH) return;
  int f = i >> 5;
  float v = (f < MF) ? proj[i] * 0.42044820762685725f : 0.0f;
  projH[i] = (_Float16)v;
}

// wcT[tap][co][ci] = conv_w[co][ci][kh][kw], tap = kh*3+kw
__global__ void convw_convert_kernel(const float* __restrict__ w,
                                     _Float16* __restrict__ wcT) {
  int i = blockIdx.x * blockDim.x + threadIdx.x;
  if (i >= 9 * CHN * CHN) return;
  int tap = i / (CHN * CHN);
  int rem = i - tap * (CHN * CHN);
  int co = rem >> 8, ci = rem & 255;
  wcT[i] = (_Float16)w[((size_t)co * CHN + ci) * 9 + tap];
}

// ---------------------------------------------------------------------------
// BatchNorm
// ---------------------------------------------------------------------------

__global__ __launch_bounds__(256)
void bn_stats_nchw_kernel(const float* __restrict__ x, float* __restrict__ sum,
                          float* __restrict__ sq) {
  int bc = blockIdx.x;
  int c = bc & (CHN - 1);
  const float* p = x + (size_t)bc * HW;
  float s = 0.f, q = 0.f;
  for (int i = threadIdx.x; i < HW; i += 256) {
    float v = p[i];
    s += v; q += v * v;
  }
  __shared__ float rs[256], rq[256];
  rs[threadIdx.x] = s; rq[threadIdx.x] = q;
  __syncthreads();
  for (int o = 128; o > 0; o >>= 1) {
    if (threadIdx.x < o) {
      rs[threadIdx.x] += rs[threadIdx.x + o];
      rq[threadIdx.x] += rq[threadIdx.x + o];
    }
    __syncthreads();
  }
  if (threadIdx.x == 0) {
    atomicAdd(&sum[c], rs[0]);
    atomicAdd(&sq[c], rq[0]);
  }
}

__global__ __launch_bounds__(256)
void bn_stats_rowmajor_kernel(const float* __restrict__ y,
                              float* __restrict__ sum, float* __restrict__ sq) {
  int c = threadIdx.x;
  size_t base = (size_t)blockIdx.x * 128 * CHN;
  float s = 0.f, q = 0.f;
  for (int r = 0; r < 128; ++r) {
    float v = y[base + (size_t)r * CHN + c];
    s += v; q += v * v;
  }
  atomicAdd(&sum[c], s);
  atomicAdd(&sq[c], q);
}

__global__ void bn_finalize_kernel(const float* __restrict__ sum,
                                   const float* __restrict__ sq,
                                   const float* __restrict__ gamma,
                                   const float* __restrict__ beta,
                                   float* __restrict__ scale,
                                   float* __restrict__ shift) {
  int c = threadIdx.x;
  if (c < CHN) {
    const float cnt = (float)NTOK;
    float m = sum[c] / cnt;
    float var = sq[c] / cnt - m * m;
    float sc = gamma[c] * rsqrtf(var + 1e-5f);
    scale[c] = sc;
    shift[c] = beta[c] - m * sc;
  }
}

// BN1 apply + ReLU, NCHW f32 -> token-major f16 via 32c x 64s LDS tile
// (both global read and write fully coalesced). grid (BATCH, CHN/32, HW/64)
__global__ __launch_bounds__(256)
void bn1_apply_kernel(const float* __restrict__ x, const float* __restrict__ scale,
                      const float* __restrict__ shift, _Float16* __restrict__ t) {
  __shared__ _Float16 tile[32][65];
  int b = blockIdx.x, c0 = blockIdx.y * 32, s0 = blockIdx.z * 64;
  for (int i = threadIdx.x; i < 2048; i += 256) {
    int c = i >> 6, s = i & 63;
    float v = x[((size_t)b * CHN + c0 + c) * HW + s0 + s];
    v = fmaxf(fmaf(scale[c0 + c], v, shift[c0 + c]), 0.0f);
    tile[c][s] = (_Float16)v;
  }
  __syncthreads();
  for (int i = threadIdx.x; i < 2048; i += 256) {
    int s = i >> 5, c = i & 31;
    t[(size_t)(b * HW + s0 + s) * 256 + c0 + c] = tile[c][s];
  }
}

// BN2 apply + ReLU, token-major f32 -> token-major f16 (fully coalesced)
__global__ __launch_bounds__(256)
void bn2_apply_kernel(const float* __restrict__ y, const float* __restrict__ scale,
                      const float* __restrict__ shift, _Float16* __restrict__ yh) {
  size_t idx = (size_t)blockIdx.x * 256 + threadIdx.x;
  int c = (int)(idx & 255);
  float v = fmaxf(fmaf(scale[c], y[idx], shift[c]), 0.0f);
  yh[idx] = (_Float16)v;
}

// ---------------------------------------------------------------------------
// Dense WMMA GEMM: C[M,256] = A[M,256] @ B[256,256] (+bias) (+NCHW residual)
// B supplied PRE-TRANSPOSED as BT[n][k]. Block tile 128x128, 8 waves
// (4 M-slices x 2 N-slices), each wave 32x64. Tiles staged with async B128
// copies into double-buffered LDS; prefetch of k+1 overlaps WMMA on k.
// mode 0: f16 out + bias (QKV).  mode 1: f32 out + bias + residual x (Wo).
// ---------------------------------------------------------------------------
__global__ __launch_bounds__(256)
void gemm_wmma_kernel(const _Float16* __restrict__ A, const _Float16* __restrict__ BT,
                      const float* __restrict__ bias, const float* __restrict__ xres,
                      _Float16* __restrict__ outH, float* __restrict__ outF,
                      int mode) {
  __shared__ _Float16 sA[2][128 * 32];
  __shared__ _Float16 sBT[2][128 * 32];
  int tid = threadIdx.x;
  int lane = tid & 31, wid = tid >> 5;
  int wm = wid & 3, wn = wid >> 2;
  int rowBase = blockIdx.x * 128;
  int colBase = blockIdx.y * 128;
  v8f acc[2][4];
#pragma unroll
  for (int i = 0; i < 2; ++i)
#pragma unroll
    for (int j = 0; j < 4; ++j) acc[i][j] = zero8();

  auto stage = [&](int buf, int k0) {
    unsigned la = lds_addr(&sA[buf][0]);
    unsigned lb = lds_addr(&sBT[buf][0]);
#pragma unroll
    for (int j = 0; j < 2; ++j) {
      int i = tid + j * 256;
      int r = i >> 2, ch = (i & 3) << 3;           // 4 x 16B chunks per 64B row
      unsigned loff = (unsigned)(r * 64 + ch * 2);
      async_b128(la + loff, A  + (size_t)(rowBase + r) * 256 + k0 + ch);
      async_b128(lb + loff, BT + (size_t)(colBase + r) * 256 + k0 + ch);
    }
  };

  stage(0, 0);
  wait_async();
  __syncthreads();

  for (int ks = 0; ks < 8; ++ks) {
    int buf = ks & 1;
    if (ks < 7) stage(buf ^ 1, (ks + 1) * 32);     // prefetch next tile
#pragma unroll
    for (int mt = 0; mt < 2; ++mt) {
      v16h af = load_frag(&sA[buf][(wm * 32 + mt * 16 + (lane & 15)) * 32], lane >> 4);
#pragma unroll
      for (int nt = 0; nt < 4; ++nt) {
        v16h bf = load_frag(&sBT[buf][(wn * 64 + nt * 16 + (lane & 15)) * 32], lane >> 4);
        acc[mt][nt] = wmma_f16(af, bf, acc[mt][nt]);
      }
    }
    wait_async();
    __syncthreads();
  }

  int nn = lane & 15, hs = lane >> 4;
#pragma unroll
  for (int mt = 0; mt < 2; ++mt)
#pragma unroll
    for (int nt = 0; nt < 4; ++nt)
#pragma unroll
      for (int r = 0; r < 8; ++r) {
        int gr = rowBase + wm * 32 + mt * 16 + r + 8 * hs;
        int gc = colBase + wn * 64 + nt * 16 + nn;
        float v = acc[mt][nt][r] + bias[gc];
        if (mode == 0) {
          outH[(size_t)gr * 256 + gc] = (_Float16)v;
        } else {
          int b = gr / HW, s = gr - b * HW;
          v += xres[((size_t)b * CHN + gc) * HW + s];
          outF[(size_t)gr * 256 + gc] = v;
        }
      }
}

// ---------------------------------------------------------------------------
// 3x3 conv as implicit GEMM (9 taps x K=256) on NHWC f16 activations.
// Weights pre-transposed wcT[tap][co][ci]; halo rows gather from a zero pad.
// Async double-buffered staging; fused residual add; NCHW f32 output.
// ---------------------------------------------------------------------------
__global__ __launch_bounds__(256)
void conv_wmma_kernel(const _Float16* __restrict__ act, const _Float16* __restrict__ wcT,
                      const _Float16* __restrict__ zpad,
                      const float* __restrict__ xres, float* __restrict__ out) {
  __shared__ _Float16 sA[2][128 * 32];
  __shared__ _Float16 sBT[2][128 * 32];
  int tid = threadIdx.x;
  int lane = tid & 31, wid = tid >> 5;
  int wm = wid & 3, wn = wid >> 2;
  int rowBase = blockIdx.x * 128;
  int colBase = blockIdx.y * 128;
  v8f acc[2][4];
#pragma unroll
  for (int i = 0; i < 2; ++i)
#pragma unroll
    for (int j = 0; j < 4; ++j) acc[i][j] = zero8();

  auto stage = [&](int buf, int step) {            // step = tap*8 + k-step
    int tap = step >> 3, k0 = (step & 7) << 5;
    int dh = tap / 3 - 1, dw = tap % 3 - 1;
    unsigned la = lds_addr(&sA[buf][0]);
    unsigned lb = lds_addr(&sBT[buf][0]);
    const _Float16* Bt = wcT + (size_t)tap * CHN * CHN;
#pragma unroll
    for (int j = 0; j < 2; ++j) {
      int i = tid + j * 256;
      int r = i >> 2, ch = (i & 3) << 3;
      int gr = rowBase + r;
      int bb = gr / HW, s = gr - bb * HW;
      int hh = s / HDIM, ww = s - hh * HDIM;
      int h2 = hh + dh, w2 = ww + dw;
      const _Float16* src =
          ((unsigned)h2 < (unsigned)HDIM && (unsigned)w2 < (unsigned)HDIM)
              ? act + (size_t)(bb * HW + h2 * HDIM + w2) * 256 + k0 + ch
              : zpad + ch;
      unsigned loff = (unsigned)(r * 64 + ch * 2);
      async_b128(la + loff, src);
      async_b128(lb + loff, Bt + (size_t)(colBase + r) * 256 + k0 + ch);
    }
  };

  stage(0, 0);
  wait_async();
  __syncthreads();

  for (int step = 0; step < 72; ++step) {
    int buf = step & 1;
    if (step < 71) stage(buf ^ 1, step + 1);
#pragma unroll
    for (int mt = 0; mt < 2; ++mt) {
      v16h af = load_frag(&sA[buf][(wm * 32 + mt * 16 + (lane & 15)) * 32], lane >> 4);
#pragma unroll
      for (int nt = 0; nt < 4; ++nt) {
        v16h bf = load_frag(&sBT[buf][(wn * 64 + nt * 16 + (lane & 15)) * 32], lane >> 4);
        acc[mt][nt] = wmma_f16(af, bf, acc[mt][nt]);
      }
    }
    wait_async();
    __syncthreads();
  }

  int nn = lane & 15, hs = lane >> 4;
#pragma unroll
  for (int mt = 0; mt < 2; ++mt)
#pragma unroll
    for (int nt = 0; nt < 4; ++nt)
#pragma unroll
      for (int r = 0; r < 8; ++r) {
        int gr = rowBase + wm * 32 + mt * 16 + r + 8 * hs;
        int gc = colBase + wn * 64 + nt * 16 + nn;
        int b = gr / HW, s = gr - b * HW;
        size_t oi = ((size_t)b * CHN + gc) * HW + s;
        out[oi] = acc[mt][nt][r] + xres[oi];
      }
}

// ---------------------------------------------------------------------------
// Performer pass A: per (b,h,split) accumulate ctx[112,32] = phi_k^T @ v and
// ksum[112] over 448 tokens (14 chunks of 32). 2 waves: wave0 owns feature
// tiles 0..3, wave1 owns 4..6.
// ---------------------------------------------------------------------------
__global__ __launch_bounds__(64)
void performer_ctx_kernel(const _Float16* __restrict__ kbuf,
                          const _Float16* __restrict__ vbuf,
                          const _Float16* __restrict__ projH,
                          float* __restrict__ ctx, float* __restrict__ ksum) {
  int blk = blockIdx.x;
  int split = blk % SPLIT_A;
  int bh = blk / SPLIT_A;
  int b = bh >> 3, h = bh & 7;
  int tid = threadIdx.x, lane = tid & 31, wid = tid >> 5;
  int ft0 = (wid == 0) ? 0 : 4;

  __shared__ _Float16 sProj[MFP * DH];   // 7 KB
  __shared__ _Float16 sPhiT[MFP * 32];   // phi transposed [feat][tok], 7 KB
  __shared__ _Float16 sVT[DH * 32];      // v transposed [d][tok], 2 KB

  for (int i = tid; i < MFP * DH / 2; i += 64)
    ((unsigned int*)sProj)[i] = ((const unsigned int*)projH)[i];
  __syncthreads();

  v16h ones;
#pragma unroll
  for (int i = 0; i < 16; ++i) ones[i] = (_Float16)1.0f;

  v8f accC[4][2], accS[4];
#pragma unroll
  for (int f = 0; f < 4; ++f) {
    accC[f][0] = zero8(); accC[f][1] = zero8(); accS[f] = zero8();
  }

  const int tokPer = HW / SPLIT_A;       // 448
  int n0base = split * tokPer;
  size_t headCol = (size_t)h * DH;

  for (int cc = 0; cc < tokPer / 32; ++cc) {
    int n0 = n0base + cc * 32;
    for (int i = tid; i < 32 * DH; i += 64) {
      int t = i >> 5, d = i & 31;
      sVT[d * 32 + t] = vbuf[(size_t)(b * HW + n0 + t) * 256 + headCol + d];
    }
    const _Float16* kbase = kbuf + (size_t)(b * HW + n0 + (lane & 15)) * 256 + headCol;
    v16h a0 = load_frag(kbase, lane >> 4);
    v16h a1 = load_frag(kbase + (size_t)16 * 256, lane >> 4);
#pragma unroll
    for (int f = 0; f < 4; ++f) {
      int ft = ft0 + f;
      if (ft < FT) {
        v16h bp = load_frag(&sProj[(ft * 16 + (lane & 15)) * DH], lane >> 4);
        v8f p0 = zero8(), p1 = zero8();
        p0 = wmma_f16(a0, bp, p0);
        p1 = wmma_f16(a1, bp, p1);
        int feat = ft * 16 + (lane & 15);
        bool fv = feat < MF;
        int hs = lane >> 4;
#pragma unroll
        for (int r = 0; r < 8; ++r) {
          int t0 = r + 8 * hs;
          sPhiT[feat * 32 + t0] =
              fv ? (_Float16)(fmaxf(p0[r], 0.f) + 1e-3f) : (_Float16)0.0f;
          sPhiT[feat * 32 + t0 + 16] =
              fv ? (_Float16)(fmaxf(p1[r], 0.f) + 1e-3f) : (_Float16)0.0f;
        }
      }
    }
    __syncthreads();
#pragma unroll
    for (int f = 0; f < 4; ++f) {
      int ft = ft0 + f;
      if (ft < FT) {
        v16h ap = load_frag(&sPhiT[(ft * 16 + (lane & 15)) * 32], lane >> 4);
        v16h b0 = load_frag(&sVT[(lane & 15) * 32], lane >> 4);
        v16h b1 = load_frag(&sVT[(16 + (lane & 15)) * 32], lane >> 4);
        accC[f][0] = wmma_f16(ap, b0, accC[f][0]);
        accC[f][1] = wmma_f16(ap, b1, accC[f][1]);
        accS[f]    = wmma_f16(ap, ones, accS[f]);
      }
    }
    __syncthreads();
  }

  int nn = lane & 15, hs = lane >> 4;
#pragma unroll
  for (int f = 0; f < 4; ++f) {
    int ft = ft0 + f;
    if (ft < FT) {
#pragma unroll
      for (int r = 0; r < 8; ++r) {
        int frow = ft * 16 + r + 8 * hs;
        atomicAdd(&ctx[((size_t)bh * MFP + frow) * DH + nn], accC[f][0][r]);
        atomicAdd(&ctx[((size_t)bh * MFP + frow) * DH + 16 + nn], accC[f][1][r]);
        if (nn == 0) atomicAdd(&ksum[(size_t)bh * MFP + frow], accS[f][r]);
      }
    }
  }
}

// ctx f32 [bh][112][32] -> ctxT f16 [bh][32][128] (feature dim zero-padded)
__global__ void ctx_finalize_kernel(const float* __restrict__ ctx,
                                    _Float16* __restrict__ ctxT) {
  size_t idx = (size_t)blockIdx.x * 256 + threadIdx.x;  // (bh*32 + d)*128 + f
  int f = (int)(idx & 127);
  int d = (int)((idx >> 7) & 31);
  int bh = (int)(idx >> 12);
  float v = (f < MFP) ? ctx[((size_t)bh * MFP + f) * DH + d] : 0.0f;
  ctxT[idx] = (_Float16)v;
}

// ---------------------------------------------------------------------------
// Performer pass B: phi_q via WMMA -> LDS; d_inv = 1/(phi_q . ksum);
// out = phi_q @ ctx * d_inv. wave0: ft 0..3 + d-tile 0; wave1: ft 4..6 + d-tile 1.
// ---------------------------------------------------------------------------
__global__ __launch_bounds__(64)
void performer_out_kernel(const _Float16* __restrict__ qbuf,
                          const _Float16* __restrict__ projH,
                          const _Float16* __restrict__ ctxT,
                          const float* __restrict__ ksum,
                          _Float16* __restrict__ attn) {
  int blk = blockIdx.x;
  int cc = blk % (HW / 32);
  int bh = blk / (HW / 32);
  int b = bh >> 3, h = bh & 7;
  int tid = threadIdx.x, lane = tid & 31, wid = tid >> 5;
  int ft0 = (wid == 0) ? 0 : 4;
  int n0 = cc * 32;

  __shared__ _Float16 sProj[MFP * DH];    // 7 KB
  __shared__ _Float16 sPhi[32 * 128];     // [tok][feat padded 128], 8 KB
  __shared__ float sKsum[MFP];
  __shared__ float sDinv[32];

  for (int i = tid; i < MFP * DH / 2; i += 64)
    ((unsigned int*)sProj)[i] = ((const unsigned int*)projH)[i];
  for (int i = tid; i < MFP; i += 64) sKsum[i] = ksum[(size_t)bh * MFP + i];
  for (int i = tid; i < 32 * 16; i += 64) {
    int t = i >> 4, f = MFP + (i & 15);
    sPhi[t * 128 + f] = (_Float16)0.0f;
  }
  __syncthreads();

  size_t headCol = (size_t)h * DH;
  const _Float16* qbase = qbuf + (size_t)(b * HW + n0 + (lane & 15)) * 256 + headCol;
  v16h a0 = load_frag(qbase, lane >> 4);
  v16h a1 = load_frag(qbase + (size_t)16 * 256, lane >> 4);
#pragma unroll
  for (int f = 0; f < 4; ++f) {
    int ft = ft0 + f;
    if (ft < FT) {
      v16h bp = load_frag(&sProj[(ft * 16 + (lane & 15)) * DH], lane >> 4);
      v8f p0 = zero8(), p1 = zero8();
      p0 = wmma_f16(a0, bp, p0);
      p1 = wmma_f16(a1, bp, p1);
      int feat = ft * 16 + (lane & 15);
      bool fv = feat < MF;
      int hs = lane >> 4;
#pragma unroll
      for (int r = 0; r < 8; ++r) {
        int t0 = r + 8 * hs;
        sPhi[t0 * 128 + feat] =
            fv ? (_Float16)(fmaxf(p0[r], 0.f) + 1e-3f) : (_Float16)0.0f;
        sPhi[(t0 + 16) * 128 + feat] =
            fv ? (_Float16)(fmaxf(p1[r], 0.f) + 1e-3f) : (_Float16)0.0f;
      }
    }
  }
  __syncthreads();

  if (wid == 0) {
    float s = 0.f;
    for (int f = 0; f < MF; ++f) s += (float)sPhi[lane * 128 + f] * sKsum[f];
    sDinv[lane] = 1.0f / s;
  }
  __syncthreads();

  v8f acc[2];
  acc[0] = zero8(); acc[1] = zero8();
  const _Float16* ctxTb = ctxT + (size_t)bh * DH * 128;
#pragma unroll
  for (int ks = 0; ks < 4; ++ks) {
    v16h bf = load_frag(&ctxTb[(size_t)(wid * 16 + (lane & 15)) * 128 + ks * 32],
                        lane >> 4);
#pragma unroll
    for (int mt = 0; mt < 2; ++mt) {
      v16h af = load_frag(&sPhi[(mt * 16 + (lane & 15)) * 128 + ks * 32], lane >> 4);
      acc[mt] = wmma_f16(af, bf, acc[mt]);
    }
  }

  int nn = lane & 15, hs = lane >> 4;
#pragma unroll
  for (int mt = 0; mt < 2; ++mt)
#pragma unroll
    for (int r = 0; r < 8; ++r) {
      int t = mt * 16 + r + 8 * hs;
      float v = acc[mt][r] * sDinv[t];
      attn[(size_t)(b * HW + n0 + t) * 256 + headCol + wid * 16 + nn] = (_Float16)v;
    }
}

// ---------------------------------------------------------------------------
// Host-side launch
// ---------------------------------------------------------------------------
extern "C" void kernel_launch(void* const* d_in, const int* in_sizes, int n_in,
                              void* d_out, int out_size, void* d_ws, size_t ws_size,
                              hipStream_t stream) {
  const float* x     = (const float*)d_in[0];
  const float* g1    = (const float*)d_in[1];
  const float* be1   = (const float*)d_in[2];
  const float* g2    = (const float*)d_in[3];
  const float* be2   = (const float*)d_in[4];
  const float* Wq    = (const float*)d_in[5];
  const float* bq    = (const float*)d_in[6];
  const float* Wk    = (const float*)d_in[7];
  const float* bk    = (const float*)d_in[8];
  const float* Wv    = (const float*)d_in[9];
  const float* bv    = (const float*)d_in[10];
  const float* Wo    = (const float*)d_in[11];
  const float* bo    = (const float*)d_in[12];
  const float* proj  = (const float*)d_in[13];
  const float* convw = (const float*)d_in[14];
  float* out = (float*)d_out;

  char* ws = (char*)d_ws;
  size_t off = 0;
  auto alloc = [&](size_t bytes) -> void* {
    off = (off + 255) & ~(size_t)255;
    void* p = ws + off;
    off += bytes;
    return p;
  };

  // Contiguous block zeroed every call: BN sums, ksum, ctx, conv zero-pad
  const int nZero = 4 * CHN + 128 * MFP + 128 * MFP * DH + 16;
  float* zblock = (float*)alloc((size_t)nZero * 4);
  float* sum1 = zblock;
  float* sq1  = zblock + CHN;
  float* sum2 = zblock + 2 * CHN;
  float* sq2  = zblock + 3 * CHN;
  float* ksum = zblock + 4 * CHN;                 // [128][112]
  float* ctx  = ksum + 128 * MFP;                 // [128][112][32]
  const _Float16* zpad = (const _Float16*)(ctx + 128 * MFP * DH);  // 64B zeros

  float* scale1 = (float*)alloc(CHN * 4);
  float* shift1 = (float*)alloc(CHN * 4);
  float* scale2 = (float*)alloc(CHN * 4);
  float* shift2 = (float*)alloc(CHN * 4);

  _Float16* projH = (_Float16*)alloc((size_t)MFP * DH * 2);
  _Float16* WqT = (_Float16*)alloc((size_t)CHN * CHN * 2);
  _Float16* WkT = (_Float16*)alloc((size_t)CHN * CHN * 2);
  _Float16* WvT = (_Float16*)alloc((size_t)CHN * CHN * 2);
  _Float16* WoT = (_Float16*)alloc((size_t)CHN * CHN * 2);
  _Float16* wcT = (_Float16*)alloc((size_t)9 * CHN * CHN * 2);
  _Float16* ctxT = (_Float16*)alloc((size_t)128 * DH * 128 * 2);

  _Float16* t    = (_Float16*)alloc((size_t)NTOK * CHN * 2);
  _Float16* q    = (_Float16*)alloc((size_t)NTOK * CHN * 2);
  _Float16* k    = (_Float16*)alloc((size_t)NTOK * CHN * 2);
  _Float16* v    = (_Float16*)alloc((size_t)NTOK * CHN * 2);
  _Float16* attn = (_Float16*)alloc((size_t)NTOK * CHN * 2);
  _Float16* yh   = (_Float16*)alloc((size_t)NTOK * CHN * 2);
  float*    y    = (float*)alloc((size_t)NTOK * CHN * 4);

  // 1) zero accumulators + weight conversions (pre-transposed for GEMM B)
  zero_f32_kernel<<<(nZero + 255) / 256, 256, 0, stream>>>(zblock, nZero);
  f32_to_f16_T_kernel<<<256, 256, 0, stream>>>(Wq, WqT);
  f32_to_f16_T_kernel<<<256, 256, 0, stream>>>(Wk, WkT);
  f32_to_f16_T_kernel<<<256, 256, 0, stream>>>(Wv, WvT);
  f32_to_f16_T_kernel<<<256, 256, 0, stream>>>(Wo, WoT);
  proj_convert_kernel<<<(MFP * DH + 255) / 256, 256, 0, stream>>>(proj, projH);
  convw_convert_kernel<<<(9 * CHN * CHN + 255) / 256, 256, 0, stream>>>(convw, wcT);

  // 2) BN1 + ReLU -> token-major f16 (LDS tile transpose)
  bn_stats_nchw_kernel<<<BATCH * CHN, 256, 0, stream>>>(x, sum1, sq1);
  bn_finalize_kernel<<<1, 256, 0, stream>>>(sum1, sq1, g1, be1, scale1, shift1);
  bn1_apply_kernel<<<dim3(BATCH, CHN / 32, HW / 64), 256, 0, stream>>>(x, scale1, shift1, t);

  // 3) QKV projections (WMMA, async-staged)
  dim3 ggrid(NTOK / 128, CHN / 128);
  gemm_wmma_kernel<<<ggrid, 256, 0, stream>>>(t, WqT, bq, nullptr, q, nullptr, 0);
  gemm_wmma_kernel<<<ggrid, 256, 0, stream>>>(t, WkT, bk, nullptr, k, nullptr, 0);
  gemm_wmma_kernel<<<ggrid, 256, 0, stream>>>(t, WvT, bv, nullptr, v, nullptr, 0);

  // 4) Performer attention
  performer_ctx_kernel<<<BATCH * HEADS * SPLIT_A, 64, 0, stream>>>(k, v, projH, ctx, ksum);
  ctx_finalize_kernel<<<(128 * DH * 128) / 256, 256, 0, stream>>>(ctx, ctxT);
  performer_out_kernel<<<BATCH * HEADS * (HW / 32), 64, 0, stream>>>(q, projH, ctxT, ksum, attn);

  // 5) Output projection + residual (f32)
  gemm_wmma_kernel<<<ggrid, 256, 0, stream>>>(attn, WoT, bo, x, nullptr, y, 1);

  // 6) BN2 + ReLU -> NHWC f16
  bn_stats_rowmajor_kernel<<<NTOK / 128, 256, 0, stream>>>(y, sum2, sq2);
  bn_finalize_kernel<<<1, 256, 0, stream>>>(sum2, sq2, g2, be2, scale2, shift2);
  bn2_apply_kernel<<<NTOK, 256, 0, stream>>>(y, scale2, shift2, yh);

  // 7) 3x3 conv (implicit GEMM, WMMA, async-staged) + residual -> d_out
  conv_wmma_kernel<<<ggrid, 256, 0, stream>>>(yh, wcT, zpad, x, out);
}
